// Attention_82652350644897
// MI455X (gfx1250) — matrix operand order repaired and checked by reading, not compiled, not run on hardware
//
#include <hip/hip_runtime.h>
#include <hip/hip_bf16.h>

typedef __bf16 bf16_t;
typedef __attribute__((ext_vector_type(16))) __bf16 v16bf;
typedef __attribute__((ext_vector_type(8)))  __bf16 v8bf;
typedef __attribute__((ext_vector_type(8)))  float  v8f;
typedef __attribute__((ext_vector_type(4)))  float  v4f;

#define NH 8

#define WMMA_BF16(a, b, c) \
  __builtin_amdgcn_wmma_f32_16x16x32_bf16(false, (a), false, (b), (short)0, (c), false, false)

static __device__ __forceinline__ v8f vzero8() {
  v8f z;
#pragma unroll
  for (int i = 0; i < 8; ++i) z[i] = 0.0f;
  return z;
}

// A-fragment (16x32 bf16, wave32): lane holds row m=lane%16;
// elements 0..7  = K: 16*hi .. 16*hi+7   (VGPRs 0..3)
// elements 8..15 = K: 16*hi+8 .. +7     (VGPRs 4..7)
static __device__ __forceinline__ v16bf afrag_bf16(const bf16_t* rowk, int hi) {
  v8bf c0 = *(const v8bf*)(rowk + 16 * hi);
  v8bf c1 = *(const v8bf*)(rowk + 16 * hi + 8);
  v16bf a;
#pragma unroll
  for (int i = 0; i < 8; ++i) { a[i] = c0[i]; a[i + 8] = c1[i]; }
  return a;
}

// ---------------------------------------------------------------------------
// Kernel A: transpose weights to bf16 so B-fragments are contiguous per lane.
// Wt[n*256 + k] = (bf16) W[k*256 + n]
// ---------------------------------------------------------------------------
__global__ __launch_bounds__(256) void prep_weights_kernel(
    const float* __restrict__ Wq, const float* __restrict__ Wk,
    const float* __restrict__ Wv, const float* __restrict__ Wg,
    const float* __restrict__ Wo,
    bf16_t* __restrict__ Wqt, bf16_t* __restrict__ Wkt,
    bf16_t* __restrict__ Wvt, bf16_t* __restrict__ Wgt,
    bf16_t* __restrict__ Wot) {
  int idx = blockIdx.x * 256 + threadIdx.x;   // 5*65536 total
  int wsel = idx >> 16;
  int r = idx & 65535;
  int k = r >> 8, n = r & 255;
  const float* src = (wsel == 0) ? Wq : (wsel == 1) ? Wk : (wsel == 2) ? Wv
                   : (wsel == 3) ? Wg : Wo;
  bf16_t* dst = (wsel == 0) ? Wqt : (wsel == 1) ? Wkt : (wsel == 2) ? Wvt
              : (wsel == 3) ? Wgt : Wot;
  dst[n * 256 + k] = (bf16_t)src[k * 256 + n];
}

// ---------------------------------------------------------------------------
// Kernel B: projection GEMMs.  M=32768, K=256, N=256.
// grid.x = M/128 row blocks, grid.y = kind (0:Q 1:K 2:V 3:gate).
// Each wave: 16 rows x 256 cols (16 accumulator tiles, 8 K-steps of 32).
// ---------------------------------------------------------------------------
__global__ __launch_bounds__(256) void proj_kernel(
    const float* __restrict__ qdata, const float* __restrict__ kdata,
    const bf16_t* __restrict__ Wqt, const bf16_t* __restrict__ Wkt,
    const bf16_t* __restrict__ Wvt, const bf16_t* __restrict__ Wgt,
    const float* __restrict__ bg,
    bf16_t* __restrict__ qh, bf16_t* __restrict__ kh,
    bf16_t* __restrict__ vtb, bf16_t* __restrict__ gate) {
  const int kind = blockIdx.y;
  const float* X = (kind == 1 || kind == 2) ? kdata : qdata;
  const bf16_t* Wt = (kind == 0) ? Wqt : (kind == 1) ? Wkt : (kind == 2) ? Wvt : Wgt;
  const int tid = threadIdx.x;
  const int w = tid >> 5, lane = tid & 31;
  const int lo = lane & 15, hi = lane >> 4;
  const int m0 = blockIdx.x * 128 + w * 16;
  const int m = m0 + lo;

  v8f acc[16];
#pragma unroll
  for (int nt = 0; nt < 16; ++nt) acc[nt] = vzero8();

#pragma unroll
  for (int kb = 0; kb < 8; ++kb) {
    // A fragment: fp32 load, convert to bf16 in registers
    const float* xr = X + (size_t)m * 256 + kb * 32;
    v4f a0 = *(const v4f*)(xr + 16 * hi);
    v4f a1 = *(const v4f*)(xr + 16 * hi + 4);
    v4f a2 = *(const v4f*)(xr + 16 * hi + 8);
    v4f a3 = *(const v4f*)(xr + 16 * hi + 12);
    v16bf a;
#pragma unroll
    for (int i = 0; i < 4; ++i) {
      a[i]      = (bf16_t)a0[i];
      a[i + 4]  = (bf16_t)a1[i];
      a[i + 8]  = (bf16_t)a2[i];
      a[i + 12] = (bf16_t)a3[i];
    }
#pragma unroll
    for (int nt = 0; nt < 16; ++nt) {
      v16bf b = *(const v16bf*)(Wt + (size_t)(nt * 16 + lo) * 256 + kb * 32 + 16 * hi);
      acc[nt] = WMMA_BF16(a, b, acc[nt]);
    }
  }

  const int s  = m0 >> 8;               // 256 rows per s (B==1)
  const int lb = (m0 & 255) + 8 * hi;   // row within s for element r: lb + r

  if (kind == 0 || kind == 1) {
    bf16_t* dst = (kind == 0) ? qh : kh;
    const float scale = (kind == 0) ? 0.1767766952966369f : 1.0f;  // d_k^-0.5
#pragma unroll
    for (int nt = 0; nt < 16; ++nt) {
      int n = nt * 16 + lo;
      int hh = n >> 5, d = n & 31;
#pragma unroll
      for (int r = 0; r < 8; ++r) {
        int l = lb + r;
        dst[(((size_t)s * NH + hh) * 256 + l) * 32 + d] = (bf16_t)(acc[nt][r] * scale);
      }
    }
  } else if (kind == 2) {
    // V stored transposed per head: [s][h][d][l]
#pragma unroll
    for (int nt = 0; nt < 16; ++nt) {
      int n = nt * 16 + lo;
      int hh = n >> 5, d = n & 31;
#pragma unroll
      for (int r = 0; r < 8; ++r) {
        int l = lb + r;
        vtb[(((size_t)s * NH + hh) * 32 + d) * 256 + l] = (bf16_t)acc[nt][r];
      }
    }
  } else {
    // gate = sigmoid(x @ Wg + bg), row-major [s*256+l][n]
#pragma unroll
    for (int nt = 0; nt < 16; ++nt) {
      int n = nt * 16 + lo;
      float bgn = bg[n];
#pragma unroll
      for (int r = 0; r < 8; ++r) {
        int l = lb + r;
        float v = acc[nt][r] + bgn;
        gate[((size_t)s * 256 + l) * 256 + n] = (bf16_t)(1.0f / (1.0f + __expf(-v)));
      }
    }
  }
}

// ---------------------------------------------------------------------------
// Kernel C: attention per (s,h).  1024 workgroups, 8 waves each.
// Wave owns 16 query rows per pass (2 passes -> 128-row blocks).
// logits: 16 WMMA (K=32).  softmax in D-layout registers with shfl_xor row
// reductions.  P staged bf16 in per-wave LDS (8KB) to re-fragment as A.
// wavg: 8 K-steps x 2 N-tiles WMMA.  Gate multiply fused on store.
// ---------------------------------------------------------------------------
__global__ __launch_bounds__(256) void attn_kernel(
    const bf16_t* __restrict__ qh, const bf16_t* __restrict__ kh,
    const bf16_t* __restrict__ vtb,
    const float* __restrict__ bias, const unsigned char* __restrict__ kmask,
    const bf16_t* __restrict__ gate, bf16_t* __restrict__ gw) {
  __shared__ bf16_t Ps[8][16 * 256];   // 64 KB: per-wave P staging
  const int sh = blockIdx.x;
  const int s = sh >> 3, h = sh & 7;
  const bf16_t* Qg = qh  + (size_t)sh * (256 * 32);
  const bf16_t* Kg = kh  + (size_t)sh * (256 * 32);
  const bf16_t* Vg = vtb + (size_t)sh * (32 * 256);
  const float* biash = bias + (size_t)h * 65536;
  const unsigned char* mrow = kmask + (size_t)s * 256;
  const int tid = threadIdx.x;
  const int w = tid >> 5, lane = tid & 31;
  const int lo = lane & 15, hi = lane >> 4;
  bf16_t* Pw = &Ps[w][0];
  const float MASKV = -3.402823466e38f;   // finfo(f32).min

  for (int mb = 0; mb < 2; ++mb) {
    const int m0 = mb * 128 + w * 16;

    // --- logits = Q K^T  (contraction d=32: one WMMA per N tile) ---
    v16bf aq = afrag_bf16(Qg + (size_t)(m0 + lo) * 32, hi);
    v8f acc[16];
#pragma unroll
    for (int nt = 0; nt < 16; ++nt) {
      v16bf bk = *(const v16bf*)(Kg + (size_t)(nt * 16 + lo) * 32 + 16 * hi);
      acc[nt] = WMMA_BF16(aq, bk, vzero8());
    }

    // --- bias + key mask + row max (lane element r is row m0+r+8*hi, col nt*16+lo) ---
    float rmax[8], rsum[8];
#pragma unroll
    for (int r = 0; r < 8; ++r) rmax[r] = MASKV;
#pragma unroll
    for (int nt = 0; nt < 16; ++nt) {
      const bool keep = mrow[nt * 16 + lo] != 0;
      const float* bp = biash + (size_t)(m0 + 8 * hi) * 256 + nt * 16 + lo;
#pragma unroll
      for (int r = 0; r < 8; ++r) {
        float v = keep ? (acc[nt][r] + bp[(size_t)r * 256]) : MASKV;
        acc[nt][r] = v;
        rmax[r] = fmaxf(rmax[r], v);
      }
    }
#pragma unroll
    for (int r = 0; r < 8; ++r) {
      float v = rmax[r];
      v = fmaxf(v, __shfl_xor(v, 1, 32));
      v = fmaxf(v, __shfl_xor(v, 2, 32));
      v = fmaxf(v, __shfl_xor(v, 4, 32));
      v = fmaxf(v, __shfl_xor(v, 8, 32));
      rmax[r] = v;
      rsum[r] = 0.0f;
    }
#pragma unroll
    for (int nt = 0; nt < 16; ++nt)
#pragma unroll
      for (int r = 0; r < 8; ++r) {
        float e = __expf(acc[nt][r] - rmax[r]);
        acc[nt][r] = e;
        rsum[r] += e;
      }
#pragma unroll
    for (int r = 0; r < 8; ++r) {
      float v = rsum[r];
      v += __shfl_xor(v, 1, 32);
      v += __shfl_xor(v, 2, 32);
      v += __shfl_xor(v, 4, 32);
      v += __shfl_xor(v, 8, 32);
      rsum[r] = 1.0f / v;
    }

    // --- stage normalized P (bf16) in this wave's LDS region ---
#pragma unroll
    for (int nt = 0; nt < 16; ++nt)
#pragma unroll
      for (int r = 0; r < 8; ++r)
        Pw[(r + 8 * hi) * 256 + nt * 16 + lo] = (bf16_t)(acc[nt][r] * rsum[r]);

    // --- wavg = P @ V  (K=256 in 8 steps, N=32 in 2 tiles) ---
    v8f wa0 = vzero8(), wa1 = vzero8();
#pragma unroll
    for (int kb = 0; kb < 8; ++kb) {
      v16bf ap = afrag_bf16(Pw + lo * 256 + kb * 32, hi);
      v16bf b0 = *(const v16bf*)(Vg + (size_t)lo * 256 + kb * 32 + 16 * hi);
      v16bf b1 = *(const v16bf*)(Vg + (size_t)(16 + lo) * 256 + kb * 32 + 16 * hi);
      wa0 = WMMA_BF16(ap, b0, wa0);
      wa1 = WMMA_BF16(ap, b1, wa1);
    }

    // --- gate multiply, store gated wavg bf16, layout [s*256+l][h*32+d] ---
    const size_t base = ((size_t)s * 256 + m0 + 8 * hi) * 256 + (size_t)h * 32;
#pragma unroll
    for (int r = 0; r < 8; ++r) {
      size_t i0 = base + (size_t)r * 256 + lo;
      size_t i1 = i0 + 16;
      gw[i0] = (bf16_t)(wa0[r] * (float)gate[i0]);
      gw[i1] = (bf16_t)(wa1[r] * (float)gate[i1]);
    }
  }
}

// ---------------------------------------------------------------------------
// Kernel D: out = gw @ Wo + bo  (fp32 output)
// ---------------------------------------------------------------------------
__global__ __launch_bounds__(256) void out_gemm_kernel(
    const bf16_t* __restrict__ gw, const bf16_t* __restrict__ Wot,
    const float* __restrict__ bo, float* __restrict__ out) {
  const int tid = threadIdx.x;
  const int w = tid >> 5, lane = tid & 31;
  const int lo = lane & 15, hi = lane >> 4;
  const int m0 = blockIdx.x * 128 + w * 16;
  const int m = m0 + lo;

  v8f acc[16];
#pragma unroll
  for (int nt = 0; nt < 16; ++nt) acc[nt] = vzero8();

#pragma unroll
  for (int kb = 0; kb < 8; ++kb) {
    v16bf a = afrag_bf16(gw + (size_t)m * 256 + kb * 32, hi);
#pragma unroll
    for (int nt = 0; nt < 16; ++nt) {
      v16bf b = *(const v16bf*)(Wot + (size_t)(nt * 16 + lo) * 256 + kb * 32 + 16 * hi);
      acc[nt] = WMMA_BF16(a, b, acc[nt]);
    }
  }
#pragma unroll
  for (int nt = 0; nt < 16; ++nt) {
    float bn = bo[nt * 16 + lo];
#pragma unroll
    for (int r = 0; r < 8; ++r)
      out[(size_t)(m0 + r + 8 * hi) * 256 + nt * 16 + lo] = acc[nt][r] + bn;
  }
}

// ---------------------------------------------------------------------------
extern "C" void kernel_launch(void* const* d_in, const int* in_sizes, int n_in,
                              void* d_out, int out_size, void* d_ws, size_t ws_size,
                              hipStream_t stream) {
  (void)in_sizes; (void)n_in; (void)out_size; (void)ws_size;
  const float* q_data = (const float*)d_in[0];
  const float* k_data = (const float*)d_in[1];
  const float* bias   = (const float*)d_in[2];
  const unsigned char* k_mask = (const unsigned char*)d_in[3];
  const float* Wq = (const float*)d_in[4];
  const float* Wk = (const float*)d_in[5];
  const float* Wv = (const float*)d_in[6];
  const float* Wg = (const float*)d_in[7];
  const float* bg = (const float*)d_in[8];
  const float* Wo = (const float*)d_in[9];
  const float* bo = (const float*)d_in[10];
  float* out = (float*)d_out;

  char* ws = (char*)d_ws;
  size_t o = 0;
  bf16_t* Wqt = (bf16_t*)(ws + o); o += 131072;
  bf16_t* Wkt = (bf16_t*)(ws + o); o += 131072;
  bf16_t* Wvt = (bf16_t*)(ws + o); o += 131072;
  bf16_t* Wgt = (bf16_t*)(ws + o); o += 131072;
  bf16_t* Wot = (bf16_t*)(ws + o); o += 131072;
  bf16_t* qhb = (bf16_t*)(ws + o); o += 16777216;   // [s][h][l][d] bf16 (scaled Q)
  bf16_t* khb = (bf16_t*)(ws + o); o += 16777216;   // [s][h][l][d]
  bf16_t* vtb = (bf16_t*)(ws + o); o += 16777216;   // [s][h][d][l]
  bf16_t* gte = (bf16_t*)(ws + o); o += 16777216;   // sigmoid gate [m][n]
  bf16_t* gwb = (bf16_t*)(ws + o); o += 16777216;   // gated wavg   [m][n]

  prep_weights_kernel<<<1280, 256, 0, stream>>>(Wq, Wk, Wv, Wg, Wo,
                                                Wqt, Wkt, Wvt, Wgt, Wot);
  dim3 gridB(256, 4);
  proj_kernel<<<gridB, 256, 0, stream>>>(q_data, k_data, Wqt, Wkt, Wvt, Wgt, bg,
                                         qhb, khb, vtb, gte);
  attn_kernel<<<1024, 256, 0, stream>>>(qhb, khb, vtb, bias, k_mask, gte, gwb);
  out_gemm_kernel<<<256, 256, 0, stream>>>(gwb, Wot, bo, out);
}
